// LPGCNHyperConvAblation_89275190215308
// MI455X (gfx1250) — compile-verified, matrix operand order checked by reading, not run on hardware
//
#include <hip/hip_runtime.h>
#include <hip/hip_bf16.h>

// ---------------------------------------------------------------------------
// CDNA5 (gfx1250) LPGCN-HyperConv pipeline.
// GEMMs: v_wmma_f32_16x16x32_bf16, weights staged in LDS in fragment order.
// Scatter phases: streaming gather + global f32 atomics (bandwidth bound).
// ---------------------------------------------------------------------------

typedef __attribute__((ext_vector_type(16))) __bf16 v16bf;
typedef __attribute__((ext_vector_type(8)))  float  v8f;

// ------------------------------ utility kernels ----------------------------

__global__ void fill_zero_f(float* __restrict__ p, long n) {
  long i = (long)blockIdx.x * blockDim.x + threadIdx.x;
  long stride = (long)gridDim.x * blockDim.x;
  for (; i < n; i += stride) p[i] = 0.0f;
}

__global__ void seg_count_k(const int* __restrict__ idx, int n, float* __restrict__ deg) {
  int i = blockIdx.x * blockDim.x + threadIdx.x;
  if (i < n) atomicAdd(&deg[idx[i]], 1.0f);
}

__global__ void invert_k(float* __restrict__ d, int n) {
  int i = blockIdx.x * blockDim.x + threadIdx.x;
  if (i < n) { float v = d[i]; d[i] = (v > 0.0f) ? (1.0f / v) : 0.0f; }
}

// GCN: deg = (#edges with dst==i) + 1 self loop; dis = rsqrt(deg)
__global__ void rsqrt1p_k(float* __restrict__ d, int n) {
  int i = blockIdx.x * blockDim.x + threadIdx.x;
  if (i < n) d[i] = rsqrtf(d[i] + 1.0f);
}

// acc[dst[p]][:] += feat[src[p]][:]   (one lane per feature dim)
__global__ void edge_scatter_k(const float* __restrict__ feat,
                               const int* __restrict__ src,
                               const int* __restrict__ dst,
                               float* __restrict__ acc, int npair, int D) {
  int p = blockIdx.x * blockDim.y + threadIdx.y;
  if (p >= npair) return;
  int d = threadIdx.x;
  int s = src[p];
  int t = dst[p];
  atomicAdd(&acc[(long)t * D + d], feat[(long)s * D + d]);
}

// data[i][d] *= s[i]
__global__ void row_scale_k(float* __restrict__ data, const float* __restrict__ s,
                            long total, int D) {
  long i = (long)blockIdx.x * blockDim.x + threadIdx.x;
  long stride = (long)gridDim.x * blockDim.x;
  for (; i < total; i += stride) {
    long row = i / D;
    data[i] *= s[row];
  }
}

// out[i][d] = acc[i][d] * s[i] + bias[d]  (optional relu)
__global__ void finish_k(float* __restrict__ out, const float* __restrict__ acc,
                         const float* __restrict__ s, const float* __restrict__ bias,
                         long total, int D, int relu) {
  long i = (long)blockIdx.x * blockDim.x + threadIdx.x;
  long stride = (long)gridDim.x * blockDim.x;
  for (; i < total; i += stride) {
    long row = i / D;
    int  d   = (int)(i - row * D);
    float v = acc[i] * s[row] + bias[d];
    if (relu) v = fmaxf(v, 0.0f);
    out[i] = v;
  }
}

// ------------------------------ WMMA GEMM ----------------------------------
// Out[M,NOUT] = concat(A0[M,K0], A1[M,K1]) @ W[K0+K1,NOUT] (+bias)
// Compile-time shapes: all tile loops unroll, bounds fold away.
// - A row index clamped (branch-free); rows >= M produce garbage only in
//   D rows >= M which are never stored.
// - K padding (K=40 case) via clamped unconditional load + select-to-zero
//   (v_cndmask), never exec-masked loads.
// - W staged once per block into LDS in per-lane WMMA fragment order (bf16),
//   so each B fragment is one contiguous 32B LDS read.
// 8 waves/block; each wave owns a 16-row strip.
template <int K0, int K1, int NOUT>
__global__ __launch_bounds__(256)
void wmma_gemm_t(const float* __restrict__ A0, const float* __restrict__ A1,
                 const float* __restrict__ W, const float* __restrict__ bias,
                 float* __restrict__ Out, int M) {
  constexpr int K   = K0 + K1;
  constexpr int NCT = (NOUT + 15) / 16;   // col tiles
  constexpr int NKT = (K + 31) / 32;      // k tiles
  constexpr int NTILES = NKT * NCT;

  // LDS: per (kt,ct) tile, per lane, 16 bf16 elements (fragment order).
  __shared__ __align__(32) __bf16 wtile[NTILES * 32 * 16];

  // ---- stage W -> LDS (bf16, fragment layout) ----
  for (int t = threadIdx.x; t < NTILES * 512; t += 256) {
    int e    = t & 15;
    int ln   = (t >> 4) & 31;
    int tile = t >> 9;
    int kt   = tile / NCT;
    int ct   = tile - kt * NCT;
    int k = kt * 32 + ((ln >> 4) << 3) + (e & 7) + ((e >= 8) ? 16 : 0);
    int n = ct * 16 + (ln & 15);
    int kc = (k < K) ? k : (K - 1);
    int nc = (n < NOUT) ? n : (NOUT - 1);
    float v = W[(long)kc * NOUT + nc];
    if (k >= K || n >= NOUT) v = 0.0f;
    wtile[t] = (__bf16)v;
  }
  __syncthreads();

  const int lane    = threadIdx.x & 31;
  const int wave    = threadIdx.x >> 5;
  const int rowBase = blockIdx.x * 128 + wave * 16;
  const int mRow    = rowBase + (lane & 15);
  const int mClamp  = (mRow < M) ? mRow : (M - 1);  // branch-free row clamp
  const int kgrp    = (lane >> 4) * 8;              // lanes 16-31 hold K+8 group
  const int ncol    = lane & 15;

  const v16bf* wfrag = (const v16bf*)wtile;

  #pragma unroll
  for (int ct = 0; ct < NCT; ++ct) {
    v8f acc = {};
    const int nb = ct * 16 + ncol;

    #pragma unroll
    for (int kt = 0; kt < NKT; ++kt) {
      const int k0 = kt * 32;

      // ---------------- A fragment (16x32 bf16) ----------------
      const float* base;
      int kk, kend, stride;
      if (k0 < K0) { base = A0; kk = k0;      kend = K0; stride = K0; }
      else         { base = A1; kk = k0 - K0; kend = K1; stride = K1; }
      const float* rowp = base + (long)mClamp * stride;
      const int kb = kk + kgrp;

      v16bf a;
      if (kk + 32 <= kend) {   // compile-time after unroll: fast path
        float4 f0 = *(const float4*)(rowp + kb);
        float4 f1 = *(const float4*)(rowp + kb + 4);
        float4 f2 = *(const float4*)(rowp + kb + 16);
        float4 f3 = *(const float4*)(rowp + kb + 20);
        a[0]  = (__bf16)f0.x; a[1]  = (__bf16)f0.y; a[2]  = (__bf16)f0.z; a[3]  = (__bf16)f0.w;
        a[4]  = (__bf16)f1.x; a[5]  = (__bf16)f1.y; a[6]  = (__bf16)f1.z; a[7]  = (__bf16)f1.w;
        a[8]  = (__bf16)f2.x; a[9]  = (__bf16)f2.y; a[10] = (__bf16)f2.z; a[11] = (__bf16)f2.w;
        a[12] = (__bf16)f3.x; a[13] = (__bf16)f3.y; a[14] = (__bf16)f3.z; a[15] = (__bf16)f3.w;
      } else {                 // K tail (K=40 head GEMM only): clamp + select
        #pragma unroll
        for (int e = 0; e < 16; ++e) {
          int k  = kb + (e & 7) + ((e >= 8) ? 16 : 0);
          int kc = (k < kend) ? k : (kend - 1);
          float v = rowp[kc];
          if (k >= kend) v = 0.0f;
          a[e] = (__bf16)v;
        }
      }

      // ---------------- B fragment: one 32B LDS read ----------------
      v16bf b = wfrag[(kt * NCT + ct) * 32 + lane];

      // EXEC all-ones here (no divergence above).
      acc = __builtin_amdgcn_wmma_f32_16x16x32_bf16(
          /*neg_a=*/false, a, /*neg_b=*/false, b,
          /*c_mod=*/(short)0, acc, /*reuse_a=*/false, /*reuse_b=*/false);
    }

    // ---------------- store D (16x16 f32) ----------------
    if (nb < NOUT) {           // folds to true for NOUT=64
      float bb = bias ? bias[nb] : 0.0f;
      if (rowBase + 16 <= M) { // uniform fast path: unconditional stores
        #pragma unroll
        for (int r = 0; r < 8; ++r) {
          int mm = rowBase + r + ((lane >> 4) << 3);
          Out[(long)mm * NOUT + nb] = acc[r] + bb;
        }
      } else {
        #pragma unroll
        for (int r = 0; r < 8; ++r) {
          int mm = rowBase + r + ((lane >> 4) << 3);
          if (mm < M) Out[(long)mm * NOUT + nb] = acc[r] + bb;
        }
      }
    }
  }
}

// ------------------------------ host helpers -------------------------------

static inline int cdiv_i(long a, long b) { return (int)((a + b - 1) / b); }

static void launch_zero(float* p, long n, hipStream_t s) {
  int blocks = cdiv_i(n, 256);
  if (blocks > 8192) blocks = 8192;
  fill_zero_f<<<blocks, 256, 0, s>>>(p, n);
}

static void launch_scatter(const float* feat, const int* src, const int* dst,
                           float* acc, int npair, int D, hipStream_t s) {
  int py = 256 / D;              // D=64 -> 4 (256 thr); D=40 -> 6 (240 thr)
  dim3 blk(D, py);
  edge_scatter_k<<<cdiv_i(npair, py), blk, 0, s>>>(feat, src, dst, acc, npair, D);
}

// ------------------------------ entry point --------------------------------

extern "C" void kernel_launch(void* const* d_in, const int* in_sizes, int n_in,
                              void* d_out, int out_size, void* d_ws, size_t ws_size,
                              hipStream_t stream) {
  (void)n_in; (void)out_size; (void)ws_size;

  const float* x    = (const float*)d_in[0];
  const int*   eidx = (const int*)d_in[1];   // [2, E]
  const int*   hidx = (const int*)d_in[2];   // [2, P]
  const float* W_h1 = (const float*)d_in[3];
  const float* b_h1 = (const float*)d_in[4];
  const float* W_h2 = (const float*)d_in[5];
  const float* b_h2 = (const float*)d_in[6];
  const float* W_c1 = (const float*)d_in[7];
  const float* b_c1 = (const float*)d_in[8];
  const float* W_c2 = (const float*)d_in[9];
  const float* b_c2 = (const float*)d_in[10];
  const float* W_lp = (const float*)d_in[11];
  const float* b_lp = (const float*)d_in[12];

  // module constants (GEMM shapes are compile-time); sizes of N/E/P from harness
  const int DIM  = 64;
  const int F_IN = 128;
  const int NC   = 40;
  const int HE   = 20000;
  const int N    = in_sizes[0] / F_IN;    // 100000
  const int E    = in_sizes[1] / 2;       // 1.6M
  const int P    = in_sizes[2] / 2;       // 800k

  const int* e_src = eidx;
  const int* e_dst = eidx + E;
  const int* n_idx = hidx;        // node side of incidence
  const int* h_idx = hidx + P;    // hyperedge side of incidence

  // ---- workspace layout (floats) ----
  float* ws    = (float*)d_ws;
  float* f_xw  = ws;                        // [N, DIM]  (xw / xcw / g1w reuse)
  float* f_acc = f_xw  + (long)N * DIM;     // [N, DIM]  node-side accumulator
  float* f_h   = f_acc + (long)N * DIM;     // [N, DIM]  h1 / x_hyper / g1 / g2
  float* a_he  = f_h   + (long)N * DIM;     // [HE, DIM] hyperedge accumulator
  float* Dinv  = a_he  + (long)HE * DIM;    // [N]
  float* Binv  = Dinv + N;                  // [HE]
  float* dis   = Binv + HE;                 // [N]

  const int gg = cdiv_i(N, 128);            // gemm grid

  // ---- degrees / normalizations (shared by both hyper convs) ----
  launch_zero(Dinv, N, stream);
  launch_zero(Binv, HE, stream);
  seg_count_k<<<cdiv_i(P, 256), 256, 0, stream>>>(n_idx, P, Dinv);
  seg_count_k<<<cdiv_i(P, 256), 256, 0, stream>>>(h_idx, P, Binv);
  invert_k<<<cdiv_i(N, 256), 256, 0, stream>>>(Dinv, N);
  invert_k<<<cdiv_i(HE, 256), 256, 0, stream>>>(Binv, HE);
  launch_zero(dis, N, stream);
  seg_count_k<<<cdiv_i(E, 256), 256, 0, stream>>>(e_dst, E, dis);
  rsqrt1p_k<<<cdiv_i(N, 256), 256, 0, stream>>>(dis, N);

  const long ND  = (long)N * DIM;
  const long HD  = (long)HE * DIM;
  const long NNC = (long)N * NC;
  const int  ew  = cdiv_i(ND, 256) > 8192 ? 8192 : cdiv_i(ND, 256);

  // ---- hypergraph conv 1: relu(D^-1 H B^-1 H^T (x W_h1) + b_h1) ----
  wmma_gemm_t<128, 0, 64><<<gg, 256, 0, stream>>>(x, nullptr, W_h1, nullptr, f_xw, N);
  launch_zero(a_he, HD, stream);
  launch_scatter(f_xw, n_idx, h_idx, a_he, P, DIM, stream);    // node -> hyperedge
  row_scale_k<<<cdiv_i(HD, 256), 256, 0, stream>>>(a_he, Binv, HD, DIM);
  launch_zero(f_acc, ND, stream);
  launch_scatter(a_he, h_idx, n_idx, f_acc, P, DIM, stream);   // hyperedge -> node
  finish_k<<<ew, 256, 0, stream>>>(f_h, f_acc, Dinv, b_h1, ND, DIM, 1);

  // ---- hypergraph conv 2 (no relu) ----
  wmma_gemm_t<64, 0, 64><<<gg, 256, 0, stream>>>(f_h, nullptr, W_h2, nullptr, f_xw, N);
  launch_zero(a_he, HD, stream);
  launch_scatter(f_xw, n_idx, h_idx, a_he, P, DIM, stream);
  row_scale_k<<<cdiv_i(HD, 256), 256, 0, stream>>>(a_he, Binv, HD, DIM);
  launch_zero(f_acc, ND, stream);
  launch_scatter(a_he, h_idx, n_idx, f_acc, P, DIM, stream);
  finish_k<<<ew, 256, 0, stream>>>(f_h, f_acc, Dinv, b_h2, ND, DIM, 0); // x_hyper

  // ---- GCN conv 1 on concat([x, x_hyper]) (concat never materialized) ----
  wmma_gemm_t<128, 64, 64><<<gg, 256, 0, stream>>>(x, f_h, W_c1, nullptr, f_xw, N);
  // y = dis * xcw ; acc init = y (self loop); acc[dst] += y[src]; out = dis*acc + b
  row_scale_k<<<ew, 256, 0, stream>>>(f_xw, dis, ND, DIM);
  hipMemcpyAsync(f_acc, f_xw, ND * sizeof(float), hipMemcpyDeviceToDevice, stream);
  launch_scatter(f_xw, e_src, e_dst, f_acc, E, DIM, stream);
  finish_k<<<ew, 256, 0, stream>>>(f_h, f_acc, dis, b_c1, ND, DIM, 1);  // g1

  // ---- GCN conv 2 (D = NC = 40, no relu) ----
  wmma_gemm_t<64, 0, 40><<<gg, 256, 0, stream>>>(f_h, nullptr, W_c2, nullptr, f_xw, N);
  row_scale_k<<<cdiv_i(NNC, 256), 256, 0, stream>>>(f_xw, dis, NNC, NC);
  hipMemcpyAsync(f_acc, f_xw, NNC * sizeof(float), hipMemcpyDeviceToDevice, stream);
  launch_scatter(f_xw, e_src, e_dst, f_acc, E, NC, stream);
  finish_k<<<cdiv_i(NNC, 256), 256, 0, stream>>>(f_h, f_acc, dis, b_c2, NNC, NC, 0); // g2

  // ---- final linear head: out = g2 @ W_lp + b_lp ----
  wmma_gemm_t<40, 0, 40><<<gg, 256, 0, stream>>>(f_h, nullptr, W_lp, b_lp, (float*)d_out, N);
}
